// SigAutoEncoder_85839216378097
// MI455X (gfx1250) — compile-verified, matrix operand order checked by reading, not run on hardware
//
#include <hip/hip_runtime.h>
#include <hip/hip_bf16.h>
#include <math.h>

typedef __attribute__((ext_vector_type(16))) _Float16 v16h;
typedef __attribute__((ext_vector_type(8)))  _Float16 v8h;
typedef __attribute__((ext_vector_type(8)))  float    v8f;

#define BB 64
#define SS 256
#define DD 512
#define HE 512
#define HD 1024

// ---- WMMA helpers -----------------------------------------------------------
// A fragment (16x32 f16): lane&15 = M row; lane<16 holds K {kb..kb+7, kb+16..kb+23},
// lane>=16 holds K {kb+8..kb+15, kb+24..kb+31}. B fragment mirrored (lane&15 = N col).
__device__ __forceinline__ v16h frag_ld(const _Float16* row, int kb, int k0) {
  union { v16h v; v8h h[2]; } u;
  u.h[0] = *(const v8h*)(row + kb + k0);
  u.h[1] = *(const v8h*)(row + kb + k0 + 16);
  return u.v;
}
__device__ __forceinline__ v8f wmma_f16(v16h a, v16h b, v8f c) {
  return __builtin_amdgcn_wmma_f32_16x16x32_f16(false, a, false, b, (short)0, c,
                                                false, false);
}
__device__ __forceinline__ float sigm(float x) { return 1.f / (1.f + __expf(-x)); }

struct EncBias { const float* bih[4]; const float* bhh[4]; };

// ---- f32 -> f16 pack --------------------------------------------------------
__global__ void cvt_f16_kernel(const float* __restrict__ s, _Float16* __restrict__ d,
                               size_t n) {
  size_t i = (size_t)blockIdx.x * blockDim.x + threadIdx.x;
  size_t st = (size_t)gridDim.x * blockDim.x;
  for (; i < n; i += st) d[i] = (_Float16)s[i];
}

// ---- z = mu + eps * exp(0.5*logvar) (also packs to f16 for decoder) ---------
__global__ void z_kernel(const float* __restrict__ mu, const float* __restrict__ lv,
                         const float* __restrict__ eps, _Float16* __restrict__ z16,
                         size_t n) {
  size_t i = (size_t)blockIdx.x * blockDim.x + threadIdx.x;
  size_t st = (size_t)gridDim.x * blockDim.x;
  for (; i < n; i += st)
    z16[i] = (_Float16)(mu[i] + eps[i] * __expf(0.5f * lv[i]));
}

// ---- sx[b,s] = x[b,s,:] . attn_W[:D] + attn_b ------------------------------
__global__ __launch_bounds__(256) void sx_kernel(const float* __restrict__ x,
                                                 const float* __restrict__ attnW,
                                                 const float* __restrict__ attnb,
                                                 float* __restrict__ sx) {
  int row  = blockIdx.x * 8 + (threadIdx.x >> 5);
  int lane = threadIdx.x & 31;
  const float* xr = x + (size_t)row * DD;
  float p = 0.f;
#pragma unroll
  for (int k = 0; k < 16; ++k) p += xr[lane + 32 * k] * attnW[lane + 32 * k];
#pragma unroll
  for (int off = 16; off; off >>= 1) p += __shfl_down(p, off, 32);
  if (lane == 0) sx[row] = p + attnb[0];
}

// ---- gi = x @ Wih.T + bih for all 4 encoder dirs (fully parallel WMMA GEMM) -
// Output gi16[dir][B*S][3*HE] (f16). One wave computes a 16x64 tile, K=512.
__global__ __launch_bounds__(256) void gi_gemm_kernel(
    const _Float16* __restrict__ x16,    // [B*S][D]
    const _Float16* __restrict__ Wih16,  // [4][3*HE][D]
    EncBias eb, _Float16* __restrict__ gi16) {
  const int lane = threadIdx.x & 31;
  const int j    = lane & 15;
  const int k0   = (lane < 16) ? 0 : 8;
  const int mhi  = (lane >> 4) << 3;
  int wtask = blockIdx.x * 8 + (threadIdx.x >> 5);  // 4 * 1024 * 24 tasks
  const int dir = wtask / (1024 * 24);
  const int rem = wtask % (1024 * 24);
  const int mt  = rem / 24;        // M tile over B*S
  const int ng  = rem % 24;        // 64-wide N group over 3*HE

  const _Float16* Wih = Wih16 + (size_t)dir * 3 * HE * DD;
  const float* bih = eb.bih[dir];
  const _Float16* xrow = x16 + (size_t)(mt * 16 + j) * DD;
  const _Float16* wrow[4];
#pragma unroll
  for (int sub = 0; sub < 4; ++sub)
    wrow[sub] = Wih + (size_t)(ng * 64 + sub * 16 + j) * DD;

  v8f acc[4] = {{}, {}, {}, {}};
#pragma unroll 2
  for (int kb = 0; kb < DD; kb += 32) {
    v16h A = frag_ld(xrow, kb, k0);
#pragma unroll
    for (int sub = 0; sub < 4; ++sub)
      acc[sub] = wmma_f16(A, frag_ld(wrow[sub], kb, k0), acc[sub]);
  }
  _Float16* go = gi16 + (size_t)dir * BB * SS * 3 * HE;
#pragma unroll
  for (int sub = 0; sub < 4; ++sub) {
    const int col = ng * 64 + sub * 16 + j;
    const float bb = bih[col];
#pragma unroll
    for (int r = 0; r < 8; ++r) {
      const int row = mt * 16 + r + mhi;
      go[(size_t)row * (3 * HE) + col] = (_Float16)(acc[sub][r] + bb);
    }
  }
}

// ---- persistent encoder GRU (one workgroup = one direction x 16 batch rows) -
// USE_GI=true: gi precomputed (bih folded in), serial step = one K=512 sweep.
template <bool USE_GI>
__global__ __launch_bounds__(1024) void enc_gru_kernel(
    const _Float16* __restrict__ x16,    // [B][S][D]
    const _Float16* __restrict__ gi16,   // [4][B*S][3*HE] or nullptr
    const _Float16* __restrict__ Wih16,  // [4][3*HE][D]
    const _Float16* __restrict__ Whh16,  // [4][3*HE][HE]
    EncBias eb, float* __restrict__ out_mu, float* __restrict__ out_lv) {
  const int lane = threadIdx.x & 31;
  const int wave = threadIdx.x >> 5;           // 0..31
  const int dir  = blockIdx.x & 3;             // 0:mu_f 1:mu_b 2:sg_f 3:sg_b
  const int b0   = (blockIdx.x >> 2) * 16;     // batch tile
  const int c    = wave * 16;                  // gate-column triplet base (HE)
  const int j    = lane & 15;
  const int k0   = (lane < 16) ? 0 : 8;
  const int mhi  = (lane >> 4) << 3;
  const bool bwd = (dir & 1);
  const int half = dir & 1;
  float* outp = (dir < 2) ? out_mu : out_lv;

  __shared__ _Float16 h16[16][520];            // hidden state, f16, padded
  for (int i = threadIdx.x; i < 16 * 520; i += 1024)
    ((_Float16*)h16)[i] = (_Float16)0.f;
  __syncthreads();

  const _Float16* Wih = Wih16 + (size_t)dir * 3 * HE * DD;
  const _Float16* Whh = Whh16 + (size_t)dir * 3 * HE * HE;
  const _Float16* gip = USE_GI ? (gi16 + (size_t)dir * BB * SS * 3 * HE) : nullptr;
  const float* bih = eb.bih[dir];
  const float* bhh = eb.bhh[dir];
  const _Float16* bi0 = Wih + (size_t)(0 * HE + c + j) * DD;
  const _Float16* bi1 = Wih + (size_t)(1 * HE + c + j) * DD;
  const _Float16* bi2 = Wih + (size_t)(2 * HE + c + j) * DD;
  const _Float16* bh0 = Whh + (size_t)(0 * HE + c + j) * HE;
  const _Float16* bh1 = Whh + (size_t)(1 * HE + c + j) * HE;
  const _Float16* bh2 = Whh + (size_t)(2 * HE + c + j) * HE;
  const float bir = bih[c + j], biz = bih[HE + c + j], bin = bih[2 * HE + c + j];
  const float bhr = bhh[c + j], bhz = bhh[HE + c + j], bhn = bhh[2 * HE + c + j];

  for (int t = 0; t < SS; ++t) {
    const int tio = bwd ? (SS - 1 - t) : t;
    v8f a0 = {}, a1 = {}, ani = {}, anh = {};
    if (!USE_GI) {
      // gi = x_t @ Wih.T   (K = D = 512), fused into serial loop
      const _Float16* xrow = x16 + ((size_t)(b0 + j) * SS + tio) * DD;
#pragma unroll 4
      for (int kb = 0; kb < DD; kb += 32) {
        v16h A = frag_ld(xrow, kb, k0);
        a0  = wmma_f16(A, frag_ld(bi0, kb, k0), a0);
        a1  = wmma_f16(A, frag_ld(bi1, kb, k0), a1);
        ani = wmma_f16(A, frag_ld(bi2, kb, k0), ani);
      }
    }
    // gh = h @ Whh.T     (K = HE = 512)
    const _Float16* hrow = &h16[j][0];
#pragma unroll 4
    for (int kb = 0; kb < HE; kb += 32) {
      v16h A = frag_ld(hrow, kb, k0);
      a0  = wmma_f16(A, frag_ld(bh0, kb, k0), a0);
      a1  = wmma_f16(A, frag_ld(bh1, kb, k0), a1);
      anh = wmma_f16(A, frag_ld(bh2, kb, k0), anh);
    }
    float hn[8];
#pragma unroll
    for (int r = 0; r < 8; ++r) {
      const int m = r + mhi;
      float gir, giz, gin;
      if (USE_GI) {
        const _Float16* g =
            gip + ((size_t)(b0 + m) * SS + tio) * (3 * HE) + c + j;
        gir = (float)g[0];
        giz = (float)g[HE];
        gin = (float)g[2 * HE];
      } else {
        gir = bir; giz = biz; gin = bin;  // gi part already in a0/a1/ani
      }
      float rg = sigm(a0[r] + gir + bhr);
      float zg = sigm(a1[r] + giz + bhz);
      float npre = USE_GI ? (gin + 0.f) : (ani[r] + gin);
      float ng = tanhf(npre + rg * (anh[r] + bhn));
      float hp = (float)h16[m][c + j];
      hn[r] = (1.f - zg) * ng + zg * hp;
    }
    __syncthreads();  // all h reads done
#pragma unroll
    for (int r = 0; r < 8; ++r) {
      const int m = r + mhi;
      h16[m][c + j] = (_Float16)hn[r];
      outp[((size_t)(b0 + m) * SS + tio) * (2 * HE) + half * HE + c + j] = hn[r];
    }
    __syncthreads();  // h writes visible
  }
}

// ---- persistent attention decoder (one workgroup = 16 batch rows) -----------
__global__ __launch_bounds__(1024) void dec_kernel(
    const _Float16* __restrict__ z16,    // [B][S][2*HE]
    const float* __restrict__ sx,        // [B*S]
    const float* __restrict__ attnW,     // [D+HD]; wah = attnW + D
    const _Float16* __restrict__ Wih16,  // [3*HD][D]
    const _Float16* __restrict__ Whh16,  // [3*HD][HD]
    const _Float16* __restrict__ lin16,  // [D][HD]
    const float* __restrict__ bih, const float* __restrict__ bhh,
    const float* __restrict__ linb, float* __restrict__ rec) {
  const int tid  = threadIdx.x;
  const int lane = tid & 31;
  const int wave = tid >> 5;     // 0..31
  const int q    = tid & 63;     // 64 threads per batch row
  const int mq   = tid >> 6;     // batch row 0..15
  const int j    = lane & 15;
  const int k0   = (lane < 16) ? 0 : 8;
  const int mhi  = (lane >> 4) << 3;
  const int b0   = blockIdx.x * 16;
  const float* wah = attnW + DD;
  const float NEGINF = -__builtin_inff();

  __shared__ _Float16 st16[16][1032];  // hidden / ctx (aliased by phase)
  __shared__ _Float16 ot16[16][520];   // previous output (x of GRU cell)
  __shared__ _Float16 w16s[16][256];   // unnormalized softmax weights
  __shared__ float red[1024];
  __shared__ float rowA[16], rowB[16], hwv[16];

  for (int i = tid; i < 16 * 1032; i += 1024) ((_Float16*)st16)[i] = (_Float16)0.f;
  for (int i = tid; i < 16 * 520;  i += 1024) ((_Float16*)ot16)[i] = (_Float16)0.f;
  __syncthreads();

  for (int i = 0; i < SS; ++i) {
    // --- phase 1: hw[b] = hidden . wah ------------------------------------
    {
      const _Float16* hr = &st16[mq][0] + q * 16;
      const float* wr = wah + q * 16;
      float p = 0.f;
#pragma unroll
      for (int k = 0; k < 16; ++k) p += (float)hr[k] * wr[k];
      red[tid] = p;
    }
    __syncthreads();
    if (tid < 16) {
      float s = 0.f;
      for (int k = 0; k < 64; ++k) s += red[tid * 64 + k];
      hwv[tid] = s;
    }
    __syncthreads();
    // --- phase 2: masked softmax ------------------------------------------
    float s4[4], mx = NEGINF;
#pragma unroll
    for (int u = 0; u < 4; ++u) {
      int s = q * 4 + u;
      float sc = sx[(size_t)(b0 + mq) * SS + s] + hwv[mq];
      if (s == i) sc = NEGINF;
      s4[u] = sc;
      mx = fmaxf(mx, sc);
    }
    red[tid] = mx;
    __syncthreads();
    if (tid < 16) {
      float m2 = NEGINF;
      for (int k = 0; k < 64; ++k) m2 = fmaxf(m2, red[tid * 64 + k]);
      rowA[tid] = m2;
    }
    __syncthreads();
    {
      float rm = rowA[mq], ps = 0.f;
#pragma unroll
      for (int u = 0; u < 4; ++u) {
        float e = __expf(s4[u] - rm);
        w16s[mq][q * 4 + u] = (_Float16)e;
        ps += e;
      }
      red[tid] = ps;
    }
    __syncthreads();
    if (tid < 16) {
      float s = 0.f;
      for (int k = 0; k < 64; ++k) s += red[tid * 64 + k];
      rowB[tid] = s;
    }
    __syncthreads();
    // --- phase 3: ctx = softmax(w) @ z  (per-row GEMV, VALU) --------------
    {
      float acc[16];
#pragma unroll
      for (int k = 0; k < 16; ++k) acc[k] = 0.f;
      const _Float16* zr = z16 + ((size_t)(b0 + mq) * SS) * (2 * HE) + q * 16;
      for (int s = 0; s < SS; ++s) {
        float e = (float)w16s[mq][s];
        v8h z0 = *(const v8h*)(zr);
        v8h z1 = *(const v8h*)(zr + 8);
#pragma unroll
        for (int k = 0; k < 8; ++k) {
          acc[k]     += e * (float)z0[k];
          acc[8 + k] += e * (float)z1[k];
        }
        zr += 2 * HE;
      }
      float inv = 1.f / rowB[mq];
#pragma unroll
      for (int k = 0; k < 16; ++k)
        st16[mq][q * 16 + k] = (_Float16)(acc[k] * inv);  // st16 now = ctx
    }
    __syncthreads();
    // --- phase 4: h_new = GRUcell(h=ctx, x=out) via WMMA ------------------
    float hnew[2][8];
#pragma unroll
    for (int task = 0; task < 2; ++task) {
      const int ct = wave + task * 32;   // 64 column tiles of HD
      const int cc = ct * 16;
      const _Float16* bi0 = Wih16 + (size_t)(0 * HD + cc + j) * DD;
      const _Float16* bi1 = Wih16 + (size_t)(1 * HD + cc + j) * DD;
      const _Float16* bi2 = Wih16 + (size_t)(2 * HD + cc + j) * DD;
      const _Float16* bh0 = Whh16 + (size_t)(0 * HD + cc + j) * HD;
      const _Float16* bh1 = Whh16 + (size_t)(1 * HD + cc + j) * HD;
      const _Float16* bh2 = Whh16 + (size_t)(2 * HD + cc + j) * HD;
      v8f a0 = {}, a1 = {}, ani = {}, anh = {};
      const _Float16* orow = &ot16[j][0];
#pragma unroll 4
      for (int kb = 0; kb < DD; kb += 32) {        // gi = out @ Wih.T
        v16h A = frag_ld(orow, kb, k0);
        a0  = wmma_f16(A, frag_ld(bi0, kb, k0), a0);
        a1  = wmma_f16(A, frag_ld(bi1, kb, k0), a1);
        ani = wmma_f16(A, frag_ld(bi2, kb, k0), ani);
      }
      const _Float16* crow = &st16[j][0];
#pragma unroll 4
      for (int kb = 0; kb < HD; kb += 32) {        // gh = ctx @ Whh.T
        v16h A = frag_ld(crow, kb, k0);
        a0  = wmma_f16(A, frag_ld(bh0, kb, k0), a0);
        a1  = wmma_f16(A, frag_ld(bh1, kb, k0), a1);
        anh = wmma_f16(A, frag_ld(bh2, kb, k0), anh);
      }
      const float bir = bih[cc + j], biz = bih[HD + cc + j], bin = bih[2 * HD + cc + j];
      const float bhr = bhh[cc + j], bhz = bhh[HD + cc + j], bhn = bhh[2 * HD + cc + j];
#pragma unroll
      for (int r = 0; r < 8; ++r) {
        const int m = r + mhi;
        float rg = sigm(a0[r] + bir + bhr);
        float zg = sigm(a1[r] + biz + bhz);
        float ng = tanhf(ani[r] + bin + rg * (anh[r] + bhn));
        float cv = (float)st16[m][cc + j];
        hnew[task][r] = (1.f - zg) * ng + zg * cv;
      }
    }
    __syncthreads();  // all ctx reads done -> safe to overwrite st16
#pragma unroll
    for (int task = 0; task < 2; ++task) {
      const int cc = (wave + task * 32) * 16;
#pragma unroll
      for (int r = 0; r < 8; ++r) st16[r + mhi][cc + j] = (_Float16)hnew[task][r];
    }
    __syncthreads();  // st16 now = hidden
    // --- phase 5: o = hidden @ lin_W.T + lin_b via WMMA -------------------
    {
      const int nn = wave * 16;  // 32 tiles of D=512
      const _Float16* lrow = lin16 + (size_t)(nn + j) * HD;
      v8f acc = {};
      const _Float16* hr = &st16[j][0];
#pragma unroll 4
      for (int kb = 0; kb < HD; kb += 32)
        acc = wmma_f16(frag_ld(hr, kb, k0), frag_ld(lrow, kb, k0), acc);
      const float bb = linb[nn + j];
#pragma unroll
      for (int r = 0; r < 8; ++r) {
        const int m = r + mhi;
        float o = acc[r] + bb;
        rec[((size_t)(b0 + m) * SS + i) * DD + nn + j] = o;
        ot16[m][nn + j] = (_Float16)o;
      }
    }
    __syncthreads();
  }
}

// ---- host launcher ----------------------------------------------------------
extern "C" void kernel_launch(void* const* d_in, const int* in_sizes, int n_in,
                              void* d_out, int out_size, void* d_ws, size_t ws_size,
                              hipStream_t stream) {
  (void)in_sizes; (void)n_in; (void)out_size;
  const float* x    = (const float*)d_in[0];
  const float* eps  = (const float*)d_in[1];
  const float* Wih[4] = {(const float*)d_in[2], (const float*)d_in[6],
                         (const float*)d_in[10], (const float*)d_in[14]};
  const float* Whh[4] = {(const float*)d_in[3], (const float*)d_in[7],
                         (const float*)d_in[11], (const float*)d_in[15]};
  EncBias eb;
  eb.bih[0] = (const float*)d_in[4];  eb.bhh[0] = (const float*)d_in[5];
  eb.bih[1] = (const float*)d_in[8];  eb.bhh[1] = (const float*)d_in[9];
  eb.bih[2] = (const float*)d_in[12]; eb.bhh[2] = (const float*)d_in[13];
  eb.bih[3] = (const float*)d_in[16]; eb.bhh[3] = (const float*)d_in[17];
  const float* dWih  = (const float*)d_in[18];
  const float* dWhh  = (const float*)d_in[19];
  const float* dbih  = (const float*)d_in[20];
  const float* dbhh  = (const float*)d_in[21];
  const float* attnW = (const float*)d_in[22];
  const float* attnb = (const float*)d_in[23];
  const float* linW  = (const float*)d_in[24];
  const float* linb  = (const float*)d_in[25];

  float* out = (float*)d_out;
  float* rec = out;                                  // [B][S][D]
  float* omu = out + (size_t)BB * SS * DD;           // [B][S][2HE]
  float* olv = omu + (size_t)BB * SS * 2 * HE;       // [B][S][2HE]

  _Float16* p = (_Float16*)d_ws;
  _Float16* x16    = p; p += (size_t)BB * SS * DD;
  _Float16* eWih16 = p; p += (size_t)4 * 3 * HE * DD;
  _Float16* eWhh16 = p; p += (size_t)4 * 3 * HE * HE;
  _Float16* dWih16 = p; p += (size_t)3 * HD * DD;
  _Float16* dWhh16 = p; p += (size_t)3 * HD * HD;
  _Float16* lin16  = p; p += (size_t)DD * HD;
  _Float16* z16    = p; p += (size_t)BB * SS * 2 * HE;
  float* sxb = (float*)p;                            // [B*S] f32
  _Float16* gi16 = (_Float16*)(sxb + (size_t)BB * SS);
  const size_t gi_elems = (size_t)4 * BB * SS * 3 * HE;  // 100.7M halves, 201MB
  const size_t base_bytes =
      ((char*)gi16) - ((char*)d_ws);
  const bool use_gi = ws_size >= base_bytes + gi_elems * sizeof(_Float16);

  auto CVT = [&](const float* s, _Float16* d, size_t n) {
    cvt_f16_kernel<<<2048, 256, 0, stream>>>(s, d, n);
  };
  CVT(x, x16, (size_t)BB * SS * DD);
  for (int d2 = 0; d2 < 4; ++d2) {
    CVT(Wih[d2], eWih16 + (size_t)d2 * 3 * HE * DD, (size_t)3 * HE * DD);
    CVT(Whh[d2], eWhh16 + (size_t)d2 * 3 * HE * HE, (size_t)3 * HE * HE);
  }
  CVT(dWih, dWih16, (size_t)3 * HD * DD);
  CVT(dWhh, dWhh16, (size_t)3 * HD * HD);
  CVT(linW, lin16, (size_t)DD * HD);

  if (use_gi) {
    // Hoist the non-recurrent input projection into a fully parallel WMMA GEMM:
    // 4 dirs x (16384 x 1536 x 512). Halves the serial per-step WMMA work.
    gi_gemm_kernel<<<12288, 256, 0, stream>>>(x16, eWih16, eb, gi16);
    enc_gru_kernel<true><<<16, 1024, 0, stream>>>(x16, gi16, eWih16, eWhh16, eb,
                                                  omu, olv);
  } else {
    enc_gru_kernel<false><<<16, 1024, 0, stream>>>(x16, nullptr, eWih16, eWhh16,
                                                   eb, omu, olv);
  }
  z_kernel<<<4096, 256, 0, stream>>>(omu, olv, eps, z16, (size_t)BB * SS * 2 * HE);
  sx_kernel<<<(BB * SS) / 8, 256, 0, stream>>>(x, attnW, attnb, sxb);
  dec_kernel<<<4, 1024, 0, stream>>>(z16, sxb, attnW, dWih16, dWhh16, lin16,
                                     dbih, dbhh, linb, rec);
}
